// RTDETRImageBranchForONNX_84499186581924
// MI455X (gfx1250) — matrix operand order repaired and checked by reading, not compiled, not run on hardware
//
#include <hip/hip_runtime.h>
#include <hip/hip_bf16.h>
#include <math.h>

// ---------------- constants ----------------
#define N_CAM 6
#define CDIM  256
#define KTOP  100
#define GKSEL 200
#define NCLS  10
#define PBIN  64
#define HW0   9216
#define HW1   2304
#define HW2   576
#define L_TOK 12096     // HW0+HW1+HW2
#define L_IMG 11520     // HW0+HW1

// ---------------- WMMA types (unique names; avoid HIP vector-type clash) ----
typedef __attribute__((ext_vector_type(16))) __bf16          v16bf;
typedef __attribute__((ext_vector_type(8)))  float           v8f;
typedef __attribute__((ext_vector_type(4)))  float           f32x4;
typedef __attribute__((ext_vector_type(4)))  unsigned short  u16x4;
typedef __attribute__((ext_vector_type(8)))  unsigned short  u16x8;

union BFrag { v16bf v; u16x8 h[2]; unsigned short u[16]; };

__device__ __forceinline__ unsigned short f2bf(float x) {
  union { float f; unsigned u; } c; c.f = x;
  unsigned r = c.u + 0x7FFFu + ((c.u >> 16) & 1u);   // round-to-nearest-even
  return (unsigned short)(r >> 16);
}

// =====================================================================
// fp32-in / bf16-WMMA / fp32-out GEMM, both operands staged via LDS.
//   TRANSA=0: A[m,k] = A[b*batchA + m*lda + k]   (row-major activations)
//   TRANSA=1: A[m,k] = A[b*batchA + k*lda + m]   (NCHW feature maps)
//   FULL=1  : M%128==0 && Kdim%32==0 && N%32==0 -> all guards compiled out
// Block 32x8 = 8 waves. Block tile: 128(M) x 32(N). Each wave: one 16-row
// strip x two 16-col WMMA accumulators (2 v_wmma per K-step, K-unroll 2).
// =====================================================================
template<int TRANSA, int FULL>
__global__ void __launch_bounds__(256) gemm_wmma_bf16(
    const float* __restrict__ A, long lda, long batchA,
    const float* __restrict__ rowscale,
    const float* __restrict__ B, int ldb,
    const float* __restrict__ bias,
    float* __restrict__ D, long ldd, long batchD,
    int M, int N, int Kdim, int relu)
{
  __shared__ __align__(16) unsigned short As[128][40];  // [m][k], padded
  __shared__ __align__(16) unsigned short Bs[32][40];   // [n][k], padded
  const float* Ab = A + (long)blockIdx.z * batchA;
  float*       Db = D + (long)blockIdx.z * batchD;
  const int lane = threadIdx.x;            // 0..31
  const int wave = threadIdx.y;            // 0..7
  const int tid  = wave * 32 + lane;
  const int m0blk = blockIdx.x * 128;
  const int n0blk = blockIdx.y * 32;

  const int mloc   = wave * 16 + (lane & 15);
  const int kbaseA = (lane >> 4) * 8;      // A frag K origin per half-wave
  const int kbaseB = (lane >> 4) * 8;      // B frag K origin per half-wave
  const int ncol   = lane & 15;

  v8f acc0 = {0.f,0.f,0.f,0.f,0.f,0.f,0.f,0.f};
  v8f acc1 = {0.f,0.f,0.f,0.f,0.f,0.f,0.f,0.f};

#pragma unroll 2
  for (int k0 = 0; k0 < Kdim; k0 += 32) {
    // ---------------- stage A tile (128 x 32) ----------------
    if (TRANSA == 0) {
      // contiguous along K: 2 threads per row, 16 K each, 4 x b128 loads
      int r = tid >> 1, kseg = (tid & 1) * 16;
      int m = m0blk + r;
      const float* src = Ab + (long)m * lda + (k0 + kseg);
      __builtin_prefetch(src + 32, 0, 1);
      if (FULL) {
        float rsr = rowscale ? rowscale[m] : 1.f;
#pragma unroll
        for (int c = 0; c < 4; ++c) {
          f32x4 v = *(const f32x4*)(src + c * 4);
          u16x4 h;
#pragma unroll
          for (int e = 0; e < 4; ++e) h[e] = f2bf(v[e] * rsr);
          *(u16x4*)&As[r][kseg + c * 4] = h;
        }
      } else {
        bool mokr = (m < M);
        float rsr = 1.f;
        if (rowscale) rsr = mokr ? rowscale[m] : 0.f;
        if (mokr && (k0 + kseg + 16 <= Kdim)) {
#pragma unroll
          for (int c = 0; c < 4; ++c) {
            f32x4 v = *(const f32x4*)(src + c * 4);
            u16x4 h;
#pragma unroll
            for (int e = 0; e < 4; ++e) h[e] = f2bf(v[e] * rsr);
            *(u16x4*)&As[r][kseg + c * 4] = h;
          }
        } else {
#pragma unroll
          for (int c = 0; c < 16; ++c) {
            int gk = k0 + kseg + c;
            float v = (mokr && gk < Kdim) ? src[c] * rsr : 0.f;
            As[r][kseg + c] = f2bf(v);
          }
        }
      }
    } else {
      // contiguous along M: 8 threads per K row, 16 M each, 4 x b128 loads
      int kk = tid >> 3, mseg = (tid & 7) * 16;
      int gk = k0 + kk;
      const float* src = Ab + (long)gk * lda + (m0blk + mseg);
      __builtin_prefetch(src + 32 * lda, 0, 1);
      if (FULL) {
#pragma unroll
        for (int c = 0; c < 4; ++c) {
          f32x4 v = *(const f32x4*)(src + c * 4);
#pragma unroll
          for (int e = 0; e < 4; ++e) As[mseg + c * 4 + e][kk] = f2bf(v[e]);
        }
      } else {
        bool kok = (gk < Kdim);
        if (kok && (m0blk + mseg + 16 <= M)) {
#pragma unroll
          for (int c = 0; c < 4; ++c) {
            f32x4 v = *(const f32x4*)(src + c * 4);
#pragma unroll
            for (int e = 0; e < 4; ++e) As[mseg + c * 4 + e][kk] = f2bf(v[e]);
          }
        } else {
#pragma unroll
          for (int c = 0; c < 16; ++c) {
            int m = m0blk + mseg + c;
            float v = (kok && m < M) ? src[c] : 0.f;
            As[mseg + c][kk] = f2bf(v);
          }
        }
      }
    }
    // ---------------- stage B tile (32K x 32N), K-major in LDS ----------
    {
      int kk = tid >> 3, nseg = (tid & 7) * 4;
      int gk = k0 + kk;
      if (FULL) {
        const float* bsrc = B + (long)gk * ldb + (n0blk + nseg);
#pragma unroll
        for (int e = 0; e < 4; ++e) Bs[nseg + e][kk] = f2bf(bsrc[e]);
      } else {
#pragma unroll
        for (int e = 0; e < 4; ++e) {
          int gn = n0blk + nseg + e;
          float v = (gk < Kdim && gn < N) ? B[(long)gk * ldb + gn] : 0.f;
          Bs[nseg + e][kk] = f2bf(v);
        }
      }
    }
    __syncthreads();

    // ---------------- fragments from LDS (pure ds_load_b128) -----------
    BFrag a, b0, b1;
    a.h[0]  = *(const u16x8*)&As[mloc][kbaseA];
    a.h[1]  = *(const u16x8*)&As[mloc][kbaseA + 16];
    b0.h[0] = *(const u16x8*)&Bs[ncol][kbaseB];
    b0.h[1] = *(const u16x8*)&Bs[ncol][kbaseB + 16];
    b1.h[0] = *(const u16x8*)&Bs[16 + ncol][kbaseB];
    b1.h[1] = *(const u16x8*)&Bs[16 + ncol][kbaseB + 16];

    acc0 = __builtin_amdgcn_wmma_f32_16x16x32_bf16(
        false, a.v, false, b0.v, (short)0, acc0, false, false);
    acc1 = __builtin_amdgcn_wmma_f32_16x16x32_bf16(
        false, a.v, false, b1.v, (short)0, acc1, false, false);
    __syncthreads();
  }

  // ---------------- epilogue ----------------
#pragma unroll
  for (int r = 0; r < 8; ++r) {
    int row = m0blk + wave * 16 + r + ((lane >> 4) * 8);
    int col0 = n0blk + ncol;
    int col1 = n0blk + 16 + ncol;
    if (FULL) {
      float v0 = acc0[r], v1 = acc1[r];
      if (bias) { v0 += bias[col0]; v1 += bias[col1]; }
      if (relu) { v0 = fmaxf(v0, 0.f); v1 = fmaxf(v1, 0.f); }
      Db[(long)row * ldd + col0] = v0;
      Db[(long)row * ldd + col1] = v1;
    } else if (row < M) {
      if (col0 < N) {
        float v = acc0[r];
        if (bias) v += bias[col0];
        if (relu) v = fmaxf(v, 0.f);
        Db[(long)row * ldd + col0] = v;
      }
      if (col1 < N) {
        float v = acc1[r];
        if (bias) v += bias[col1];
        if (relu) v = fmaxf(v, 0.f);
        Db[(long)row * ldd + col1] = v;
      }
    }
  }
}

// ---------------- LayerNorm over rows of 256 ----------------
__global__ void __launch_bounds__(256) layernorm_rows(
    float* __restrict__ X, const float* __restrict__ g,
    const float* __restrict__ b)
{
  int row = blockIdx.x, t = threadIdx.x;
  __shared__ float red[256];
  float x = X[(long)row * CDIM + t];
  red[t] = x; __syncthreads();
  for (int off = 128; off; off >>= 1) { if (t < off) red[t] += red[t + off]; __syncthreads(); }
  float mean = red[0] * (1.0f / CDIM); __syncthreads();
  float dx = x - mean;
  red[t] = dx * dx; __syncthreads();
  for (int off = 128; off; off >>= 1) { if (t < off) red[t] += red[t + off]; __syncthreads(); }
  float var = red[0] * (1.0f / CDIM);
  X[(long)row * CDIM + t] = dx * rsqrtf(var + 1e-5f) * g[t] + b[t];
}

// ---------------- row max over 10 classes ----------------
__global__ void rowmax_cls(const float* __restrict__ cls,
                           float* __restrict__ s, int rows)
{
  int r = blockIdx.x * blockDim.x + threadIdx.x;
  if (r >= rows) return;
  float m = cls[(long)r * NCLS];
  for (int n = 1; n < NCLS; ++n) m = fmaxf(m, cls[(long)r * NCLS + n]);
  s[r] = m;
}

// ---------------- analytic anchors ----------------
__device__ __forceinline__ void anchor_logits(int l, float a[4])
{
  int lvl, h, w, off;
  if (l < HW0)           { lvl = 0; h = 72; w = 128; off = 0; }
  else if (l < HW0 + HW1){ lvl = 1; h = 36; w = 64;  off = HW0; }
  else                   { lvl = 2; h = 18; w = 32;  off = HW0 + HW1; }
  int li = l - off, gy = li / w, gx = li % w;
  float c[4];
  c[0] = (gx + 0.5f) / (float)w;
  c[1] = (gy + 0.5f) / (float)h;
  c[2] = 0.05f * (float)(1 << lvl);
  c[3] = c[2];
  bool valid = true;
#pragma unroll
  for (int i = 0; i < 4; ++i) valid = valid && (c[i] > 0.01f) && (c[i] < 0.99f);
#pragma unroll
  for (int i = 0; i < 4; ++i) a[i] = valid ? logf(c[i] / (1.f - c[i])) : INFINITY;
}

__global__ void make_vmask(float* __restrict__ vmask)
{
  int i = blockIdx.x * blockDim.x + threadIdx.x;
  if (i >= N_CAM * L_TOK) return;
  float a[4]; anchor_logits(i % L_TOK, a);
  vmask[i] = isinf(a[0]) ? 0.f : 1.f;
}

// ---------------- deterministic iterative top-k ----------------
__global__ void __launch_bounds__(256) topk_percam(
    float* __restrict__ scores, float* __restrict__ tks, int* __restrict__ tki)
{
  int cam = blockIdx.x, t = threadIdx.x;
  float* s = scores + (long)cam * L_TOK;
  __shared__ float bv[256]; __shared__ int bi[256];
  for (int r = 0; r < KTOP; ++r) {
    float best = -INFINITY; int bidx = 0x7fffffff;
    for (int i = t; i < L_TOK; i += 256) {
      float v = s[i];
      if (v > best || (v == best && i < bidx)) { best = v; bidx = i; }
    }
    bv[t] = best; bi[t] = bidx; __syncthreads();
    for (int off = 128; off; off >>= 1) {
      if (t < off) {
        if (bv[t + off] > bv[t] || (bv[t + off] == bv[t] && bi[t + off] < bi[t]))
        { bv[t] = bv[t + off]; bi[t] = bi[t + off]; }
      }
      __syncthreads();
    }
    if (t == 0) { tks[cam * KTOP + r] = bv[0]; tki[cam * KTOP + r] = bi[0]; s[bi[0]] = -INFINITY; }
    __syncthreads();
  }
}

__global__ void __launch_bounds__(256) topk_global(
    float* __restrict__ tks, int* __restrict__ gidx)
{
  int t = threadIdx.x;
  __shared__ float bv[256]; __shared__ int bi[256];
  for (int r = 0; r < GKSEL; ++r) {
    float best = -INFINITY; int bidx = 0x7fffffff;
    for (int i = t; i < N_CAM * KTOP; i += 256) {
      float v = tks[i];
      if (v > best || (v == best && i < bidx)) { best = v; bidx = i; }
    }
    bv[t] = best; bi[t] = bidx; __syncthreads();
    for (int off = 128; off; off >>= 1) {
      if (t < off) {
        if (bv[t + off] > bv[t] || (bv[t + off] == bv[t] && bi[t + off] < bi[t]))
        { bv[t] = bv[t + off]; bi[t] = bi[t + off]; }
      }
      __syncthreads();
    }
    if (t == 0) { gidx[r] = bi[0]; tks[bi[0]] = -INFINITY; }
    __syncthreads();
  }
}

// ---------------- gathers / small heads ----------------
__global__ void gather_qf(const float* __restrict__ out_mem,
                          const int* __restrict__ tki, float* __restrict__ qf)
{
  int q = blockIdx.x, c = threadIdx.x;
  int cam = q / KTOP;
  qf[(long)q * CDIM + c] = out_mem[((long)cam * L_TOK + tki[q]) * CDIM + c];
}

__global__ void make_refxy(const float* __restrict__ bbraw,
                           const int* __restrict__ tki, float* __restrict__ refxy)
{
  int q = blockIdx.x * blockDim.x + threadIdx.x;
  if (q >= N_CAM * KTOP) return;
  float a[4]; anchor_logits(tki[q], a);
  float x = bbraw[q * 4 + 0] + a[0];
  float y = bbraw[q * 4 + 1] + a[1];
  refxy[q * 2 + 0] = 1.f / (1.f + expf(-x));
  refxy[q * 2 + 1] = 1.f / (1.f + expf(-y));
}

__global__ void make_depth_in(const float* __restrict__ qf,
                              const int* __restrict__ gidx,
                              const float* __restrict__ intr,
                              float* __restrict__ din)
{
  int g = blockIdx.x, c = threadIdx.x;
  if (c >= CDIM + 16) return;
  int q = gidx[g], cam = q / KTOP;
  float v = (c < CDIM) ? qf[(long)q * CDIM + c]
                       : intr[cam * 16 + (c - CDIM)] * 0.01f;
  din[(long)g * (CDIM + 16) + c] = v;
}

__global__ void softmax64(float* __restrict__ dl)
{
  int g = blockIdx.x, d = threadIdx.x;
  __shared__ float red[64];
  float x = dl[g * PBIN + d];
  if (isnan(x)) x = 0.f;
  else if (isinf(x)) x = (x > 0.f) ? 1e4f : -1e4f;
  red[d] = x; __syncthreads();
  for (int off = 32; off; off >>= 1) { if (d < off) red[d] = fmaxf(red[d], red[d + off]); __syncthreads(); }
  float mx = red[0]; __syncthreads();
  float e = expf(x - mx);
  red[d] = e; __syncthreads();
  for (int off = 32; off; off >>= 1) { if (d < off) red[d] += red[d + off]; __syncthreads(); }
  dl[g * PBIN + d] = e / red[0];
}

__global__ void make_dyn(const int* __restrict__ gidx,
                         const float* __restrict__ refxy,
                         const float* __restrict__ intr,
                         const float* __restrict__ extinv,
                         const float* __restrict__ dbins,
                         const float* __restrict__ prob,
                         float* __restrict__ out)
{
  int g = blockIdx.x, d = threadIdx.x;
  int q = gidx[g], cam = q / KTOP;
  float fx = fmaxf(intr[cam * 16 + 0], 1e-6f);
  float fy = fmaxf(intr[cam * 16 + 5], 1e-6f);
  float cx = intr[cam * 16 + 2], cy = intr[cam * 16 + 6];
  float u = refxy[q * 2 + 0] * 1024.f, v = refxy[q * 2 + 1] * 576.f;
  float depth = dbins[d];
  float pt[4] = { (u - cx) / fx * depth, (v - cy) / fy * depth, depth, 1.f };
  long base = ((long)g * PBIN + d) * 4;
#pragma unroll
  for (int j = 0; j < 3; ++j) {
    float s = 0.f;
#pragma unroll
    for (int k = 0; k < 4; ++k) s += pt[k] * extinv[cam * 16 + j * 4 + k];
    out[base + j] = s;
  }
  out[base + 3] = prob[g * PBIN + d];
}

__global__ void write_qfout(const float* __restrict__ qf,
                            const int* __restrict__ gidx, float* __restrict__ out)
{
  int g = blockIdx.x, c = threadIdx.x;
  out[(long)g * CDIM + c] = qf[(long)gidx[g] * CDIM + c];
}

// ---------------- MLN gamma/beta (tiny) ----------------
__global__ void __launch_bounds__(256) mln_gamma_beta(
    const float* __restrict__ intr, const float* __restrict__ ext,
    const float* __restrict__ w_m1, const float* __restrict__ b_m1,
    const float* __restrict__ w_mg, const float* __restrict__ b_mg,
    const float* __restrict__ w_mb, const float* __restrict__ b_mb,
    float* __restrict__ gamma, float* __restrict__ beta)
{
  int cam = blockIdx.x, c = threadIdx.x;
  __shared__ float hm[CDIM];
  __shared__ float in14[14];
  if (c == 0) { in14[0] = intr[cam * 16 + 0] * 0.001f; in14[1] = intr[cam * 16 + 5] * 0.001f; }
  if (c >= 2 && c < 14) in14[c] = ext[cam * 16 + (c - 2)];
  __syncthreads();
  float acc = b_m1[c];
  for (int i = 0; i < 14; ++i) acc += in14[i] * w_m1[i * CDIM + c];
  hm[c] = fmaxf(acc, 0.f);
  __syncthreads();
  float ga = b_mg[c], be = b_mb[c];
  for (int j = 0; j < CDIM; ++j) { float h = hm[j]; ga += h * w_mg[j * CDIM + c]; be += h * w_mb[j * CDIM + c]; }
  gamma[cam * CDIM + c] = ga;
  beta[cam * CDIM + c]  = be;
}

// -------- feat_flatten: LDS 32x32 tile transpose + affine --------
__global__ void __launch_bounds__(256) feat_mln(
    const float* __restrict__ feat0, const float* __restrict__ feat1,
    const float* __restrict__ gamma, const float* __restrict__ beta,
    float* __restrict__ out)
{
  __shared__ float tile[32][33];
  int cam = blockIdx.z;
  int l0 = blockIdx.x * 32;
  int c0 = blockIdx.y * 32;
  int tx = threadIdx.x, ty = threadIdx.y;   // 32 x 8
  const float* src; int lbase, hw;
  if (l0 < HW0) { src = feat0; lbase = 0;   hw = HW0; }
  else          { src = feat1; lbase = HW0; hw = HW1; }
  int ll = l0 - lbase + tx;
#pragma unroll
  for (int i = 0; i < 4; ++i) {
    int cl = ty + i * 8;
    tile[cl][tx] = src[((long)cam * CDIM + (c0 + cl)) * hw + ll]; // coalesced over HW
  }
  __syncthreads();
  int c = c0 + tx;
  float ga = gamma[cam * CDIM + c], be = beta[cam * CDIM + c];
#pragma unroll
  for (int i = 0; i < 4; ++i) {
    int l = l0 + ty + i * 8;
    out[((long)cam * L_IMG + l) * CDIM + c] = tile[tx][ty + i * 8] * ga + be; // coalesced over C
  }
}

// ---------------- lidar2img + spatial constants ----------------
__global__ void make_l2i(const float* __restrict__ intr,
                         const float* __restrict__ ext, float* __restrict__ out)
{
  int idx = threadIdx.x;
  if (idx >= 96) return;
  int n = idx / 16, r = (idx % 16) / 4, cc = idx % 4;
  float s = 0.f;
  for (int j = 0; j < 4; ++j) s += intr[n * 16 + r * 4 + j] * ext[n * 16 + j * 4 + cc];
  out[idx] = s;
}

__global__ void write_spat(float* out_f)
{
  if (threadIdx.x == 0) {
    int* o = (int*)out_f;
    o[0] = 72; o[1] = 128; o[2] = 36; o[3] = 64;
  }
}

// =====================================================================
// kernel_launch
// =====================================================================
extern "C" void kernel_launch(void* const* d_in, const int* in_sizes, int n_in,
                              void* d_out, int out_size, void* d_ws, size_t ws_size,
                              hipStream_t stream)
{
  (void)in_sizes; (void)n_in; (void)out_size; (void)ws_size;
  const float* feat0  = (const float*)d_in[0];
  const float* feat1  = (const float*)d_in[1];
  const float* feat2  = (const float*)d_in[2];
  const float* intr   = (const float*)d_in[3];
  const float* ext    = (const float*)d_in[4];
  const float* extinv = (const float*)d_in[5];
  const float* w_p0   = (const float*)d_in[6];
  const float* w_p1   = (const float*)d_in[7];
  const float* w_p2   = (const float*)d_in[8];
  const float* w_enc  = (const float*)d_in[9];
  const float* b_enc  = (const float*)d_in[10];
  const float* ln_g   = (const float*)d_in[11];
  const float* ln_b   = (const float*)d_in[12];
  const float* w_cls  = (const float*)d_in[13];
  const float* b_cls  = (const float*)d_in[14];
  const float* w_bb1  = (const float*)d_in[15];
  const float* b_bb1  = (const float*)d_in[16];
  const float* w_bb2  = (const float*)d_in[17];
  const float* b_bb2  = (const float*)d_in[18];
  const float* w_bb3  = (const float*)d_in[19];
  const float* b_bb3  = (const float*)d_in[20];
  const float* dbins  = (const float*)d_in[21];
  const float* w_d1   = (const float*)d_in[22];
  const float* b_d1   = (const float*)d_in[23];
  const float* w_d2   = (const float*)d_in[24];
  const float* b_d2   = (const float*)d_in[25];
  const float* w_m1   = (const float*)d_in[26];
  const float* b_m1   = (const float*)d_in[27];
  const float* w_mg   = (const float*)d_in[28];
  const float* b_mg   = (const float*)d_in[29];
  const float* w_mb   = (const float*)d_in[30];
  const float* b_mb   = (const float*)d_in[31];

  float* out = (float*)d_out;
  const long FEAT_O = 0;                                   // 6*11520*256
  const long SPAT_O = FEAT_O + (long)N_CAM * L_IMG * CDIM; // 4 ints
  const long L2I_O  = SPAT_O + 4;                          // 96
  const long DYN_O  = L2I_O + 96;                          // 200*64*4
  const long QFO_O  = DYN_O + (long)GKSEL * PBIN * 4;      // 200*256

  float* ws = (float*)d_ws;
  const long MEM_O   = 0;
  const long OUTM_O  = MEM_O  + (long)N_CAM * L_TOK * CDIM;
  const long CLS_O   = OUTM_O + (long)N_CAM * L_TOK * CDIM;
  const long VMSK_O  = CLS_O  + (long)N_CAM * L_TOK * NCLS;
  const long SCR_O   = VMSK_O + (long)N_CAM * L_TOK;
  const long TKS_O   = SCR_O  + (long)N_CAM * L_TOK;
  const long TKI_O   = TKS_O  + N_CAM * KTOP;
  const long QF_O    = TKI_O  + N_CAM * KTOP;
  const long H1_O    = QF_O   + (long)N_CAM * KTOP * CDIM;
  const long H2_O    = H1_O   + (long)N_CAM * KTOP * CDIM;
  const long BBR_O   = H2_O   + (long)N_CAM * KTOP * CDIM;
  const long RXY_O   = BBR_O  + N_CAM * KTOP * 4;
  const long GIDX_O  = RXY_O  + N_CAM * KTOP * 2;
  const long DIN_O   = GIDX_O + GKSEL;
  const long HD_O    = DIN_O  + (long)GKSEL * (CDIM + 16);
  const long DL_O    = HD_O   + (long)GKSEL * CDIM;
  const long GAM_O   = DL_O   + (long)GKSEL * PBIN;
  const long BET_O   = GAM_O  + N_CAM * CDIM;

  float* mem   = ws + MEM_O;
  float* outm  = ws + OUTM_O;
  float* clsb  = ws + CLS_O;
  float* vmask = ws + VMSK_O;
  float* scr   = ws + SCR_O;
  float* tks   = ws + TKS_O;
  int*   tki   = (int*)(ws + TKI_O);
  float* qf    = ws + QF_O;
  float* h1    = ws + H1_O;
  float* h2    = ws + H2_O;
  float* bbraw = ws + BBR_O;
  float* refxy = ws + RXY_O;
  int*   gidx  = (int*)(ws + GIDX_O);
  float* din   = ws + DIN_O;
  float* hd    = ws + HD_O;
  float* dl    = ws + DL_O;
  float* gam   = ws + GAM_O;
  float* bet   = ws + BET_O;

  const dim3 gblk(32, 8);
  const int MTOT = N_CAM * L_TOK;   // 72576 = 567 * 128

  make_vmask<<<(MTOT + 255) / 256, 256, 0, stream>>>(vmask);

  // 1x1 input projections (A = feat^T per camera; batched over cameras)
  gemm_wmma_bf16<1,1><<<dim3(HW0 / 128, 8, N_CAM), gblk, 0, stream>>>(
      feat0, (long)HW0, (long)CDIM * HW0, nullptr, w_p0, CDIM, nullptr,
      mem, CDIM, (long)L_TOK * CDIM, HW0, CDIM, CDIM, 0);
  gemm_wmma_bf16<1,1><<<dim3(HW1 / 128, 8, N_CAM), gblk, 0, stream>>>(
      feat1, (long)HW1, (long)CDIM * HW1, nullptr, w_p1, CDIM, nullptr,
      mem + (long)HW0 * CDIM, CDIM, (long)L_TOK * CDIM, HW1, CDIM, CDIM, 0);
  gemm_wmma_bf16<1,0><<<dim3((HW2 + 127) / 128, 8, N_CAM), gblk, 0, stream>>>(
      feat2, (long)HW2, (long)CDIM * HW2, nullptr, w_p2, CDIM, nullptr,
      mem + (long)(HW0 + HW1) * CDIM, CDIM, (long)L_TOK * CDIM, HW2, CDIM, CDIM, 0);

  // encoder: (memory * valid) @ w_enc + b_enc, then LN
  gemm_wmma_bf16<0,1><<<dim3(MTOT / 128, 8, 1), gblk, 0, stream>>>(
      mem, (long)CDIM, 0L, vmask, w_enc, CDIM, b_enc,
      outm, CDIM, 0L, MTOT, CDIM, CDIM, 0);
  layernorm_rows<<<MTOT, 256, 0, stream>>>(outm, ln_g, ln_b);

  // classification logits + per-token max score
  gemm_wmma_bf16<0,0><<<dim3(MTOT / 128, 1, 1), gblk, 0, stream>>>(
      outm, (long)CDIM, 0L, nullptr, w_cls, NCLS, b_cls,
      clsb, NCLS, 0L, MTOT, NCLS, CDIM, 0);
  rowmax_cls<<<(MTOT + 255) / 256, 256, 0, stream>>>(clsb, scr, MTOT);

  // per-camera top-100
  topk_percam<<<N_CAM, 256, 0, stream>>>(scr, tks, tki);

  // gather selected encoder tokens
  gather_qf<<<N_CAM * KTOP, CDIM, 0, stream>>>(outm, tki, qf);

  // bbox MLP on the 600 selected rows only
  gemm_wmma_bf16<0,0><<<dim3(5, 8, 1), gblk, 0, stream>>>(
      qf, (long)CDIM, 0L, nullptr, w_bb1, CDIM, b_bb1,
      h1, CDIM, 0L, N_CAM * KTOP, CDIM, CDIM, 1);
  gemm_wmma_bf16<0,0><<<dim3(5, 8, 1), gblk, 0, stream>>>(
      h1, (long)CDIM, 0L, nullptr, w_bb2, CDIM, b_bb2,
      h2, CDIM, 0L, N_CAM * KTOP, CDIM, CDIM, 1);
  gemm_wmma_bf16<0,0><<<dim3(5, 1, 1), gblk, 0, stream>>>(
      h2, (long)CDIM, 0L, nullptr, w_bb3, 4, b_bb3,
      bbraw, 4, 0L, N_CAM * KTOP, 4, CDIM, 0);
  make_refxy<<<3, 256, 0, stream>>>(bbraw, tki, refxy);

  // global top-200
  topk_global<<<1, 256, 0, stream>>>(tks, gidx);

  // depth head on the 200 selected rows
  make_depth_in<<<GKSEL, 288, 0, stream>>>(qf, gidx, intr, din);
  gemm_wmma_bf16<0,0><<<dim3(2, 8, 1), gblk, 0, stream>>>(
      din, (long)(CDIM + 16), 0L, nullptr, w_d1, CDIM, b_d1,
      hd, CDIM, 0L, GKSEL, CDIM, CDIM + 16, 1);
  gemm_wmma_bf16<0,0><<<dim3(2, 2, 1), gblk, 0, stream>>>(
      hd, (long)CDIM, 0L, nullptr, w_d2, PBIN, b_d2,
      dl, PBIN, 0L, GKSEL, PBIN, CDIM, 0);
  softmax64<<<GKSEL, PBIN, 0, stream>>>(dl);

  // dyn_query + qf_out
  make_dyn<<<GKSEL, PBIN, 0, stream>>>(gidx, refxy, intr, extinv, dbins, dl,
                                       out + DYN_O);
  write_qfout<<<GKSEL, CDIM, 0, stream>>>(qf, gidx, out + QFO_O);

  // MLN affine + big transpose output
  mln_gamma_beta<<<N_CAM, 256, 0, stream>>>(intr, ext, w_m1, b_m1,
                                            w_mg, b_mg, w_mb, b_mb, gam, bet);
  feat_mln<<<dim3(L_IMG / 32, CDIM / 32, N_CAM), dim3(32, 8), 0, stream>>>(
      feat0, feat1, gam, bet, out + FEAT_O);

  // lidar2img + spatial constants
  make_l2i<<<1, 128, 0, stream>>>(intr, ext, out + L2I_O);
  write_spat<<<1, 32, 0, stream>>>(out + SPAT_O);
}
